// graphAttentionHead_58205396795605
// MI455X (gfx1250) — matrix operand order repaired and checked by reading, not compiled, not run on hardware
//
#include <hip/hip_runtime.h>

typedef __attribute__((ext_vector_type(16))) _Float16 v16h;
typedef __attribute__((ext_vector_type(2)))  _Float16 v2h;
typedef __attribute__((ext_vector_type(8)))  float    v8f;
typedef __attribute__((ext_vector_type(4)))  float    v4f;
typedef __attribute__((ext_vector_type(4)))  unsigned v4u;
typedef __attribute__((ext_vector_type(8)))  unsigned v8u;

#define NN 8192
#define DIM 128
#define NEG_INF -9.0e15f
#define ALPHA 0.2f
#define THRESH 0.5f

__device__ __forceinline__ float gat_score(float d, float s2v, float s1v) {
  float x  = s1v + s2v;
  float lr = fmaxf(x, ALPHA * x);          // leaky_relu
  return d < THRESH ? lr : NEG_INF;        // distance mask
}

// ---------- kernel 1: s1 = nodes@a1, s2 = nodes@a2 (one wave per row) ----------
__global__ void gat_scores_k(const float* __restrict__ nodes,
                             const float* __restrict__ a,
                             float* __restrict__ s1,
                             float* __restrict__ s2) {
  int wid  = (int)((blockIdx.x * blockDim.x + threadIdx.x) >> 5);  // row 0..8191
  int lane = threadIdx.x & 31;
  const v4f* nr = (const v4f*)(nodes + (size_t)wid * DIM);
  const v4f* a1 = (const v4f*)(a);
  const v4f* a2 = (const v4f*)(a + DIM);
  v4f x = nr[lane], b1 = a1[lane], b2 = a2[lane];
  float p1 = x.x*b1.x + x.y*b1.y + x.z*b1.z + x.w*b1.w;
  float p2 = x.x*b2.x + x.y*b2.y + x.z*b2.z + x.w*b2.w;
  #pragma unroll
  for (int off = 16; off >= 1; off >>= 1) {
    p1 += __shfl_xor(p1, off, 32);
    p2 += __shfl_xor(p2, off, 32);
  }
  if (lane == 0) { s1[wid] = p1; s2[wid] = p2; }
}

// ---------- kernel 2: pack nodes (f32) into f16 WMMA-B operand layout ----------
// VB[(kb*8+t)*64 + chunk*32 + lane] : v4u = dwords i (4i..4i+3), dword i holds
// half2( nodes[kb*32 + 2i + 8*(i>=4) + 8*(lane>=16)][t*16+lane%16], nodes[k+1][same col] )
__global__ void gat_pack_k(const float* __restrict__ nodes,
                           unsigned* __restrict__ vb) {
  int wid  = (int)((blockIdx.x * blockDim.x + threadIdx.x) >> 5);  // 0..2047
  int lane = threadIdx.x & 31;
  int kb = wid >> 3, t = wid & 7;
  int hi = lane >> 4;
  int c  = t * 16 + (lane & 15);
  int k0 = kb * 32 + 8 * hi;
  unsigned o[8];
  #pragma unroll
  for (int i = 0; i < 8; ++i) {
    v2h p;
    p.x = (_Float16)nodes[(size_t)(k0 + 2*i + 8*(i>>2)    ) * DIM + c];
    p.y = (_Float16)nodes[(size_t)(k0 + 2*i + 8*(i>>2) + 1) * DIM + c];
    o[i] = __builtin_bit_cast(unsigned, p);
  }
  v4u* out4 = (v4u*)vb;
  size_t base = (size_t)(kb * 8 + t) * 64;
  v4u lo4, hi4;
  lo4.x = o[0]; lo4.y = o[1]; lo4.z = o[2]; lo4.w = o[3];
  hi4.x = o[4]; hi4.y = o[5]; hi4.z = o[6]; hi4.w = o[7];
  out4[base + lane]      = lo4;
  out4[base + 32 + lane] = hi4;
}

// ---------- kernel 3: flash-attention main loop (one wave = 16 rows x 128 cols) ----
__global__ void __launch_bounds__(256) gat_main_k(
    const float* __restrict__ dist,
    const float* __restrict__ s1,
    const float* __restrict__ s2,
    const unsigned* __restrict__ vb,
    float* __restrict__ pm, float* __restrict__ pl, float* __restrict__ pacc,
    int S) {
  int wid  = (int)((blockIdx.x * blockDim.x + threadIdx.x) >> 5);  // 0..512*S-1
  int lane = threadIdx.x & 31;
  int M  = lane & 15, hi = lane >> 4;
  int s  = wid >> 9;          // k-split index
  int w  = wid & 511;         // row-block index
  int i0 = w << 4;
  int kspan = NN / S;
  int kbeg  = s * kspan;

  float s1v = s1[i0 + M];
  float m = NEG_INF, l = 0.0f;
  v8f acc[8];
  #pragma unroll
  for (int t = 0; t < 8; ++t)
    #pragma unroll
    for (int j = 0; j < 8; ++j) acc[t][j] = 0.0f;

  const v4u*  vb4   = (const v4u*)vb;
  const float* dbase = dist + (size_t)(i0 + M) * NN + 8 * hi;  // A-layout row/lane

  for (int k0 = kbeg; k0 < kbeg + kspan; k0 += 32) {
    // dist tile, loaded directly in WMMA-A (16x32 f16) lane layout; NT: stream-once
    const v4f* drow = (const v4f*)(dbase + k0);
    v4f d0 = __builtin_nontemporal_load(drow + 0);   // K rel +0..3
    v4f d1 = __builtin_nontemporal_load(drow + 1);   // K rel +4..7
    v4f d2 = __builtin_nontemporal_load(drow + 4);   // K rel +16..19
    v4f d3 = __builtin_nontemporal_load(drow + 5);   // K rel +20..23
    const v4f* s2p = (const v4f*)(s2 + k0 + 8 * hi);
    v4f t0 = s2p[0], t1 = s2p[1], t2 = s2p[4], t3 = s2p[5];

    float sc[16];
    sc[0] = gat_score(d0.x, t0.x, s1v); sc[1] = gat_score(d0.y, t0.y, s1v);
    sc[2] = gat_score(d0.z, t0.z, s1v); sc[3] = gat_score(d0.w, t0.w, s1v);
    sc[4] = gat_score(d1.x, t1.x, s1v); sc[5] = gat_score(d1.y, t1.y, s1v);
    sc[6] = gat_score(d1.z, t1.z, s1v); sc[7] = gat_score(d1.w, t1.w, s1v);
    sc[8] = gat_score(d2.x, t2.x, s1v); sc[9] = gat_score(d2.y, t2.y, s1v);
    sc[10]= gat_score(d2.z, t2.z, s1v); sc[11]= gat_score(d2.w, t2.w, s1v);
    sc[12]= gat_score(d3.x, t3.x, s1v); sc[13]= gat_score(d3.y, t3.y, s1v);
    sc[14]= gat_score(d3.z, t3.z, s1v); sc[15]= gat_score(d3.w, t3.w, s1v);

    // online softmax: row M lives in lanes M and M+16
    float mx = sc[0];
    #pragma unroll
    for (int j = 1; j < 16; ++j) mx = fmaxf(mx, sc[j]);
    mx = fmaxf(mx, __shfl_xor(mx, 16, 32));
    float mn = fmaxf(m, mx);
    if (__any(mn > m)) {                       // wave-uniform: EXEC stays full
      float scale = __expf(m - mn);            // exp(-inf)=0 handles first tile
      l *= scale;
      m  = mn;
      float sj[8];
      #pragma unroll
      for (int j = 0; j < 8; ++j) sj[j] = __shfl(scale, hi * 8 + j, 32);
      #pragma unroll
      for (int t = 0; t < 8; ++t)
        #pragma unroll
        for (int j = 0; j < 8; ++j) acc[t][j] *= sj[j];
    }

    float rs = 0.0f;
    v16h A;
    #pragma unroll
    for (int j = 0; j < 16; ++j) {
      float p = __expf(sc[j] - m);
      rs += p;
      A[j] = (_Float16)p;                      // element order == A-operand K order
    }
    rs += __shfl_xor(rs, 16, 32);
    l += rs;

    // 8 x v_wmma_f32_16x16x32_f16, B pre-packed -> 2 coalesced b128 loads each
    size_t bbase = (size_t)(k0 >> 5) * 512;
    #pragma unroll
    for (int t = 0; t < 8; ++t) {
      size_t o = bbase + (size_t)t * 64;
      v8u bq;
      bq.lo = vb4[o + lane];
      bq.hi = vb4[o + 32 + lane];
      v16h B = __builtin_bit_cast(v16h, bq);
      acc[t] = __builtin_amdgcn_wmma_f32_16x16x32_f16(
                   false, A, false, B, (short)0, acc[t], false, false);
    }
  }

  // write per-split partials (unnormalized)
  if (hi == 0) {
    pm[(size_t)s * NN + i0 + M] = m;
    pl[(size_t)s * NN + i0 + M] = l;
  }
  #pragma unroll
  for (int t = 0; t < 8; ++t)
    #pragma unroll
    for (int j = 0; j < 8; ++j)    // C/D layout: lane holds rows hi*8+j, col lane%16
      pacc[((size_t)s * NN + i0 + hi * 8 + j) * DIM + t * 16 + M] = acc[t][j];
}

// ---------- kernel 4: merge k-splits (log-sum-exp) + normalize ----------
__global__ void gat_combine_k(const float* __restrict__ pm,
                              const float* __restrict__ pl,
                              const float* __restrict__ pacc,
                              float* __restrict__ out, int S) {
  int idx = blockIdx.x * blockDim.x + threadIdx.x;   // 0..NN*DIM-1
  int row = idx >> 7;
  float ms = pm[row];
  for (int s = 1; s < S; ++s) ms = fmaxf(ms, pm[s * NN + row]);
  float num = 0.0f, den = 0.0f;
  for (int s = 0; s < S; ++s) {
    float wgt = __expf(pm[s * NN + row] - ms);
    num += wgt * pacc[(size_t)s * NN * DIM + idx];
    den += wgt * pl[s * NN + row];
  }
  out[idx] = num / den;
}

extern "C" void kernel_launch(void* const* d_in, const int* in_sizes, int n_in,
                              void* d_out, int out_size, void* d_ws, size_t ws_size,
                              hipStream_t stream) {
  (void)in_sizes; (void)n_in; (void)out_size;
  const float* nodes = (const float*)d_in[0];
  const float* dist  = (const float*)d_in[1];
  const float* a     = (const float*)d_in[2];
  float* out = (float*)d_out;

  // workspace layout
  float*    s1 = (float*)d_ws;
  float*    s2 = s1 + NN;
  unsigned* vb = (unsigned*)(s2 + NN);          // 524288 dwords = 2 MB
  float*    pm = (float*)(vb + 524288);

  // pick k-split factor S (power of two) from available scratch
  size_t fixed = (size_t)(2 * NN) * 4 + (size_t)524288 * 4;
  size_t per   = ((size_t)NN * 2 + (size_t)NN * DIM) * 4;   // m,l,acc per split
  int S = 1;
  if      (ws_size >= fixed + 8 * per) S = 8;
  else if (ws_size >= fixed + 4 * per) S = 4;
  else if (ws_size >= fixed + 2 * per) S = 2;

  float* pl   = pm + (size_t)S * NN;
  float* pacc = pl + (size_t)S * NN;

  gat_scores_k<<<NN / 8, 256, 0, stream>>>(nodes, a, s1, s2);
  gat_pack_k  <<<256,    256, 0, stream>>>(nodes, vb);
  gat_main_k  <<<64 * S, 256, 0, stream>>>(dist, s1, s2, vb, pm, pl, pacc, S);
  gat_combine_k<<<(NN * DIM) / 256, 256, 0, stream>>>(pm, pl, pacc, out, S);
}
